// BasicBlock_24352464569360
// MI455X (gfx1250) — 8-bit matrix operand order repaired and checked by reading, compiled, not run on hardware
//
#include <hip/hip_runtime.h>
#include <cstdint>
#include <cstddef>

typedef int   v8i  __attribute__((ext_vector_type(8)));
typedef int   v16i __attribute__((ext_vector_type(16)));
typedef float v8f  __attribute__((ext_vector_type(8)));

#define C_IN   128
#define HH     56
#define WW     56
#define BB     32
#define HP     58
#define WP     58
#define NPIX   (BB*HH*WW)
#define NTILES (NPIX/16)

#define FP8_P1 0x38
#define FP8_M1 0xB8
#define FP8_Z  0x00

#define OFF_XBIN  ((size_t)0)
#define SZ_XBIN   ((size_t)BB*HP*WP*C_IN)
#define OFF_BP1   (OFF_XBIN + SZ_XBIN)
#define SZ_BP     ((size_t)8*9*32*16*4)
#define OFF_BP2   (OFF_BP1 + SZ_BP)
#define OFF_TMP   (OFF_BP2 + SZ_BP)
#define SZ_TMP    ((size_t)NPIX*C_IN*4)
#define OFF_STATS (OFF_TMP + SZ_TMP)
#define OFF_SB    (OFF_STATS + 512*4)

#define BP_DWORDS (8*9*32*16)
#define LDS_BYTES (BP_DWORDS*4 + 2*C_IN*4)

__device__ __forceinline__ uint8_t fp8sign(float v) {
    return (v > 0.f) ? (uint8_t)FP8_P1 : ((v < 0.f) ? (uint8_t)FP8_M1 : (uint8_t)FP8_Z);
}

__global__ void prep_kernel(const float* __restrict__ x, uint8_t* __restrict__ xbin,
                            float* __restrict__ stats) {
    int idx = blockIdx.x * 256 + threadIdx.x;
    if (idx < 512) stats[idx] = 0.f;
    int c  = idx & (C_IN - 1);
    int r  = idx >> 7;
    int xx = r % WP;
    int r2 = r / WP;
    int y  = r2 % HP;
    int b  = r2 / HP;
    uint8_t v = FP8_Z;
    if (y >= 1 && y <= HH && xx >= 1 && xx <= WW) {
        float f = x[(((size_t)(b * C_IN + c)) * HH + (y - 1)) * WW + (xx - 1)];
        v = fp8sign(f);
    }
    xbin[idx] = v;
}

__global__ void pack_weights_kernel(const float* __restrict__ w1, const float* __restrict__ w2,
                                    int* __restrict__ bp1, int* __restrict__ bp2) {
    int idx  = blockIdx.x * 256 + threadIdx.x;
    int conv = idx / 36864;
    int r    = idx % 36864;
    const float* w = conv ? w2 : w1;
    int* bp        = conv ? bp2 : bp1;

    int vv    = r & 3;
    int lane  = (r >> 2) & 31;
    int rd    = (r >> 7) & 3;
    int nk    = r >> 9;
    int kk    = nk % 9;
    int ntile = nk / 9;

    int n    = ntile * 16 + (lane & 15);
    int half = lane >> 4;
    int kl0  = rd * 32 + (vv >> 1) * 16 + half * 8 + (vv & 1) * 4;

    unsigned word = 0;
    #pragma unroll
    for (int j = 0; j < 4; ++j) {
        int c = kl0 + j;
        float wv = w[((size_t)(n * C_IN + c)) * 9 + kk];
        word |= ((unsigned)fp8sign(wv)) << (8 * j);
    }
    bp[r] = (int)word;
}

__global__ __launch_bounds__(256) void conv_bin_kernel(
    const uint8_t* __restrict__ abin,
    const int*     __restrict__ bpack,
    float*         __restrict__ outbuf,
    const float*   __restrict__ residual,
    float* __restrict__ gsum, float* __restrict__ gsq) {

    extern __shared__ char smem[];
    int*   bsh  = (int*)smem;
    float* lsum = (float*)(smem + BP_DWORDS * 4);
    float* lsq  = lsum + C_IN;

    int tid = threadIdx.x;

    {
        const int4* src = (const int4*)bpack;
        int4*       dst = (int4*)bsh;
        #pragma unroll
        for (int i = 0; i < 36; ++i)
            dst[i * 256 + tid] = src[i * 256 + tid];
    }
    if (tid < C_IN) { lsum[tid] = 0.f; lsq[tid] = 0.f; }
    __syncthreads();

    int wave  = tid >> 5;
    int lane  = tid & 31;
    int half  = lane >> 4;
    int m     = lane & 15;
    int mtile = blockIdx.x * 8 + wave;

    int p  = mtile * 16 + m;
    int b  = p / (HH * WW);
    int hw = p % (HH * WW);
    int h  = hw / WW, w = hw % WW;
    const uint8_t* abase = abin + (((size_t)(b * HP + h)) * WP + w) * C_IN + half * 8;
    const int*     fbase = bsh + lane * 4;

    auto loadA = [&](int tap) -> v16i {
        int kh = tap / 3, kw = tap - kh * 3;
        const uint8_t* ap = abase + ((kh * WP + kw) * C_IN);
        unsigned long long q0 = *(const unsigned long long*)(ap + 0);
        unsigned long long q1 = *(const unsigned long long*)(ap + 16);
        unsigned long long q2 = *(const unsigned long long*)(ap + 32);
        unsigned long long q3 = *(const unsigned long long*)(ap + 48);
        unsigned long long q4 = *(const unsigned long long*)(ap + 64);
        unsigned long long q5 = *(const unsigned long long*)(ap + 80);
        unsigned long long q6 = *(const unsigned long long*)(ap + 96);
        unsigned long long q7 = *(const unsigned long long*)(ap + 112);
        v16i a;
        a[0]  = (int)q0; a[1]  = (int)(q0 >> 32);
        a[2]  = (int)q1; a[3]  = (int)(q1 >> 32);
        a[4]  = (int)q2; a[5]  = (int)(q2 >> 32);
        a[6]  = (int)q3; a[7]  = (int)(q3 >> 32);
        a[8]  = (int)q4; a[9]  = (int)(q4 >> 32);
        a[10] = (int)q5; a[11] = (int)(q5 >> 32);
        a[12] = (int)q6; a[13] = (int)(q6 >> 32);
        a[14] = (int)q7; a[15] = (int)(q7 >> 32);
        return a;
    };

    auto loadB = [&](int kkx, int t) -> v16i {
        const int* ft = fbase + kkx * 512 + t * (9 * 512);
        int4 b0 = *(const int4*)(ft);
        int4 b1 = *(const int4*)(ft + 128);
        int4 b2 = *(const int4*)(ft + 256);
        int4 b3 = *(const int4*)(ft + 384);
        v16i bf;
        bf[0]  = b0.x; bf[1]  = b0.y; bf[2]  = b0.z; bf[3]  = b0.w;
        bf[4]  = b1.x; bf[5]  = b1.y; bf[6]  = b1.z; bf[7]  = b1.w;
        bf[8]  = b2.x; bf[9]  = b2.y; bf[10] = b2.z; bf[11] = b2.w;
        bf[12] = b3.x; bf[13] = b3.y; bf[14] = b3.z; bf[15] = b3.w;
        return bf;
    };

    v8f acc[8];
    v8f zero = {0.f, 0.f, 0.f, 0.f, 0.f, 0.f, 0.f, 0.f};
    #pragma unroll
    for (int t = 0; t < 8; ++t) acc[t] = zero;

    #pragma unroll
    for (int kk = 0; kk < 9; ++kk) {
        v16i a = loadA(kk);
        #pragma unroll
        for (int t = 0; t < 8; ++t) {
            v16i bf = loadB(kk, t);
            acc[t] = __builtin_amdgcn_wmma_f32_16x16x128_fp8_fp8(a, bf, (short)0, acc[t],
                                                                 false, false);
        }
    }

    __builtin_amdgcn_sched_group_barrier(0x100, 8, 0);
    #pragma unroll
    for (int i = 0; i < 70; ++i) {
        __builtin_amdgcn_sched_group_barrier(0x008, 1, 0);
        __builtin_amdgcn_sched_group_barrier(0x100, 4, 0);
    }
    __builtin_amdgcn_sched_group_barrier(0x008, 1, 0);
    __builtin_amdgcn_sched_group_barrier(0x008, 1, 0);

    #pragma unroll
    for (int t = 0; t < 8; ++t) {
        int n = t * 16 + (lane & 15);
        float s1 = 0.f, s2 = 0.f;
        #pragma unroll
        for (int r = 0; r < 8; ++r) {
            int mloc = r + half * 8;
            int pp   = mtile * 16 + mloc;
            float f  = acc[t][r];
            if (residual != nullptr) {
                int bb  = pp / (HH * WW);
                int hw2 = pp % (HH * WW);
                f += residual[((size_t)(bb * C_IN + n)) * (HH * WW) + hw2];
            }
            outbuf[(size_t)pp * C_IN + n] = f;
            s1 += f;
            s2 += f * f;
        }
        atomicAdd(&lsum[n], s1);
        atomicAdd(&lsq[n], s2);
    }
    __syncthreads();
    if (tid < C_IN) {
        atomicAdd(&gsum[tid], lsum[tid]);
        atomicAdd(&gsq[tid], lsq[tid]);
    }
}

__global__ void bn_finalize_kernel(const float* __restrict__ sum, const float* __restrict__ sq,
                                   const float* __restrict__ gamma, const float* __restrict__ beta,
                                   float* __restrict__ scale, float* __restrict__ bias) {
    int c = threadIdx.x;
    const float invM = 1.0f / (float)NPIX;
    float mean = sum[c] * invM;
    float var  = sq[c] * invM - mean * mean;
    float sc   = gamma[c] * rsqrtf(var + 1e-5f);
    scale[c] = sc;
    bias[c]  = beta[c] - mean * sc;
}

__global__ void apply_bn_sign_kernel(const float* __restrict__ tmp,
                                     const float* __restrict__ scale, const float* __restrict__ bias,
                                     uint8_t* __restrict__ actbin) {
    int idx = blockIdx.x * 256 + threadIdx.x;
    int c = idx & (C_IN - 1);
    int p = idx >> 7;
    float f = tmp[idx] * scale[c] + bias[c];
    int b  = p / (HH * WW);
    int hw = p % (HH * WW);
    int h  = hw / WW, w = hw % WW;
    actbin[(((size_t)(b * HP + (h + 1))) * WP + (w + 1)) * C_IN + c] = fp8sign(f);
}

__global__ void final_out_kernel(const float* __restrict__ tmp,
                                 const float* __restrict__ scale, const float* __restrict__ bias,
                                 float* __restrict__ out) {
    int idx = blockIdx.x * 256 + threadIdx.x;
    int hw = idx % (HH * WW);
    int bc = idx / (HH * WW);
    int c  = bc & (C_IN - 1);
    int b  = bc >> 7;
    int p  = b * (HH * WW) + hw;
    float f = tmp[(size_t)p * C_IN + c];
    f = f * scale[c] + bias[c];
    f = fminf(1.0f, fmaxf(-1.0f, f));
    out[idx] = f;
}

extern "C" void kernel_launch(void* const* d_in, const int* in_sizes, int n_in,
                              void* d_out, int out_size, void* d_ws, size_t ws_size,
                              hipStream_t stream) {
    const float* x  = (const float*)d_in[0];
    const float* w1 = (const float*)d_in[1];
    const float* w2 = (const float*)d_in[2];
    const float* g1 = (const float*)d_in[3];
    const float* b1 = (const float*)d_in[4];
    const float* g2 = (const float*)d_in[5];
    const float* b2 = (const float*)d_in[6];

    char*    ws    = (char*)d_ws;
    uint8_t* xbin  = (uint8_t*)(ws + OFF_XBIN);
    int*     bp1   = (int*)(ws + OFF_BP1);
    int*     bp2   = (int*)(ws + OFF_BP2);
    float*   tmp   = (float*)(ws + OFF_TMP);
    float*   stats = (float*)(ws + OFF_STATS);
    float*   sb    = (float*)(ws + OFF_SB);

    prep_kernel<<<(BB * HP * WP * C_IN) / 256, 256, 0, stream>>>(x, xbin, stats);
    pack_weights_kernel<<<(2 * 36864) / 256, 256, 0, stream>>>(w1, w2, bp1, bp2);
    conv_bin_kernel<<<NTILES / 8, 256, LDS_BYTES, stream>>>(xbin, bp1, tmp, nullptr,
                                                            stats + 0, stats + 128);
    bn_finalize_kernel<<<1, 128, 0, stream>>>(stats + 0, stats + 128, g1, b1, sb + 0, sb + 128);
    apply_bn_sign_kernel<<<(NPIX * C_IN) / 256, 256, 0, stream>>>(tmp, sb + 0, sb + 128, xbin);
    conv_bin_kernel<<<NTILES / 8, 256, LDS_BYTES, stream>>>(xbin, bp2, tmp, x,
                                                            stats + 256, stats + 384);
    bn_finalize_kernel<<<1, 128, 0, stream>>>(stats + 256, stats + 384, g2, b2, sb + 256, sb + 384);
    final_out_kernel<<<(NPIX * C_IN) / 256, 256, 0, stream>>>(tmp, sb + 256, sb + 384,
                                                              (float*)d_out);
}
